// H3Conv_68032281968966
// MI455X (gfx1250) — matrix operand order repaired and checked by reading, not compiled, not run on hardware
//
#include <hip/hip_runtime.h>
#include <hip/hip_bf16.h>
#include <cstdint>
#include <cstddef>

// ---- geometry (matches reference) ----
#define BB 4
#define LL 2048
#define DM 1024
#define LPAD (LL + 32)   // reversed kernel + causal zero pad

typedef __attribute__((ext_vector_type(16))) __bf16        v16bf;
typedef __attribute__((ext_vector_type(8)))  float         v8f;
typedef __attribute__((ext_vector_type(4)))  unsigned int  u32x4;
typedef __attribute__((ext_vector_type(4)))  unsigned int  v4u;
typedef __attribute__((ext_vector_type(8)))  int           v8i;
typedef __attribute__((ext_vector_type(4)))  int           v4i;

// Tensor Data Mover availability (this toolchain: clang-23 6-arg form, device only)
#if defined(__gfx1250__) && __has_builtin(__builtin_amdgcn_tensor_load_to_lds) && \
    __has_builtin(__builtin_amdgcn_s_wait_tensorcnt)
#define USE_TDM 1
#else
#define USE_TDM 0
#endif

union Frag {
  u32x4        q[2];
  unsigned int u[8];
  v16bf        bf;
};

__device__ __forceinline__ unsigned short f2bf(float f) {
  unsigned int x = __float_as_uint(f);
  x += 0x7FFFu + ((x >> 16) & 1u);     // round-to-nearest-even
  return (unsigned short)(x >> 16);
}

__device__ __forceinline__ v8f wmma_bf16(v16bf a, v16bf b, v8f c) {
  // v_wmma_f32_16x16x32_bf16: D = A(16x32) * B(32x16) + C(16x16 f32)
  return __builtin_amdgcn_wmma_f32_16x16x32_bf16(
      /*neg_a*/false, a, /*neg_b*/false, b,
      /*c_mod*/(short)0, c, /*reuse_a*/false, /*reuse_b*/false);
}

// ---------------- fp32 -> bf16 conversion pass ----------------
__global__ __launch_bounds__(256) void cvt_bf16(const float* __restrict__ src,
                                                unsigned short* __restrict__ dst,
                                                int n) {
  int i = blockIdx.x * 256 + threadIdx.x;
  if (i < n) dst[i] = f2bf(src[i]);
}

// ---------------- bf16 WMMA GEMM:  C(MxN) = A(MxK) * W(NxK)^T + bias ----------------
// One wave per 32x32 C tile. K = DM = 1024.
// mode 0: store transposed into (b, d, l) fp32 buffer   (QKV projections)
//         + optional bf16 shadow copy (for TDM staging of the conv input)
// mode 1: store row-major (bl, d) fp32 buffer           (output projection)
__global__ __launch_bounds__(32) void gemm_bf16(
    const unsigned short* __restrict__ A,   // (M, K) bf16 row-major
    const unsigned short* __restrict__ W,   // (N, K) bf16 row-major
    const float* __restrict__ bias,         // (N)
    float* __restrict__ out,
    unsigned short* __restrict__ out_bf,    // optional bf16 copy (mode 0 only)
    int mode)
{
  const int K = DM;
  const int lane = threadIdx.x;
  const int r = lane & 15;            // row (A) / col (B) within tile
  const int h = lane >> 4;            // half-wave selector
  const int m0 = blockIdx.x << 5;
  const int n0 = blockIdx.y << 5;

  const v8f vzero = {0.f,0.f,0.f,0.f,0.f,0.f,0.f,0.f};
  v8f acc00 = vzero, acc01 = vzero, acc10 = vzero, acc11 = vzero;

  // A fragment per-lane base: row = m0 + 16*mi + r ; 16 contiguous K at k0 + h*16
  const unsigned short* pa0 = A + (size_t)(m0 + r) * K + (h << 4);
  const unsigned short* pa1 = pa0 + (size_t)16 * K;
  // B fragment per-lane base: row(d) = n0 + 16*ni + r ; 16 contiguous K at k0 + h*16
  const unsigned short* pb0 = W + (size_t)(n0 + r) * K + (h << 4);
  const unsigned short* pb1 = pb0 + (size_t)16 * K;

  for (int k0 = 0; k0 < K; k0 += 32) {
    Frag a0, a1, b0, b1;
    a0.q[0] = *(const u32x4*)(pa0 + k0);
    a0.q[1] = *(const u32x4*)(pa0 + k0 + 8);
    a1.q[0] = *(const u32x4*)(pa1 + k0);
    a1.q[1] = *(const u32x4*)(pa1 + k0 + 8);
    b0.q[0] = *(const u32x4*)(pb0 + k0);
    b0.q[1] = *(const u32x4*)(pb0 + k0 + 8);
    b1.q[0] = *(const u32x4*)(pb1 + k0);
    b1.q[1] = *(const u32x4*)(pb1 + k0 + 8);
    __builtin_prefetch(pa0 + k0 + 32, 0, 1);   // global_prefetch_b8
    __builtin_prefetch(pb0 + k0 + 32, 0, 1);

    acc00 = wmma_bf16(a0.bf, b0.bf, acc00);
    acc01 = wmma_bf16(a0.bf, b1.bf, acc01);
    acc10 = wmma_bf16(a1.bf, b0.bf, acc10);
    acc11 = wmma_bf16(a1.bf, b1.bf, acc11);
  }

  v8f accs[2][2] = {{acc00, acc01}, {acc10, acc11}};
  #pragma unroll
  for (int ni = 0; ni < 2; ++ni) {
    const int d = n0 + (ni << 4) + r;
    const float bs = bias[d];
    #pragma unroll
    for (int mi = 0; mi < 2; ++mi) {
      const int mbase = m0 + (mi << 4) + (h << 3);   // 8 consecutive M rows per lane
      v8f c = accs[mi][ni];
      if (mode == 0) {
        const int b = mbase >> 11;                   // L = 2048
        const int l = mbase & (LL - 1);
        const size_t off = (((size_t)(b << 10) + d) << 11) + l;  // (b*DM+d)*L + l
        float* p = out + off;
        #pragma unroll
        for (int v = 0; v < 8; ++v) p[v] = c[v] + bs;
        if (out_bf) {
          unsigned short* pb = out_bf + off;
          #pragma unroll
          for (int v = 0; v < 8; ++v) pb[v] = f2bf(c[v] + bs);
        }
      } else {
        #pragma unroll
        for (int v = 0; v < 8; ++v)
          out[(size_t)(mbase + v) * DM + d] = c[v] + bs;
      }
    }
  }
}

// ---------------- per-channel causal conv via Toeplitz WMMA ----------------
// y[l,b] = sum_m ker[d, l-m] * x[m,b]  ==  A(Toeplitz of ker) * X
// One block per channel d; 8 waves each own 16 output tiles of 16 along L.
// X rows (bf16, 4 batches x 2048) are staged to LDS with one TENSOR_LOAD_TO_LDS.
// mode 0: out = (conv(k) + k_D[d]*k) * v   -> fp32 + bf16 (b,d,l)  [alias X]
// mode 1: out = (conv(kv) + D[d]*kv) * q   -> bf16 (b*L+l, d) transposed
__global__ __launch_bounds__(256) void conv_gate(
    const float* __restrict__ ker,            // (DM, L) fp32
    const float* __restrict__ X,              // (B, DM, L) fp32 conv input
    const unsigned short* __restrict__ Xbf,   // (B, DM, L) bf16 conv input
    const float* __restrict__ skip,           // (DM)
    const float* __restrict__ gate,           // (B, DM, L) fp32
    float* __restrict__ outA,                 // mode 0 fp32 destination
    unsigned short* __restrict__ outAbf,      // mode 0 bf16 destination
    unsigned short* __restrict__ outB,        // mode 1 bf16 destination
    int mode)
{
  // rker32[i] packs (rker[i], rker[i+1]) so any fragment dword is ONE ds_load_b32.
  __shared__ __align__(16) unsigned int   rker32[LPAD];
  __shared__ __align__(16) unsigned short ldsx[5 * LL];   // 4 batch rows + zero row

  const int d   = blockIdx.x;
  const int tid = threadIdx.x;

  // Stage reversed bf16 kernel pairs: rker[i] = ker[d][L-1-i], zeros past L.
  for (int i = tid; i < LPAD; i += 256) {
    unsigned int lo = 0, hi = 0;
    if (i < LL)     lo = f2bf(ker[(size_t)d * LL + (LL - 1 - i)]);
    if (i + 1 < LL) hi = f2bf(ker[(size_t)d * LL + (LL - 2 - i)]);
    rker32[i] = lo | (hi << 16);
  }

#if USE_TDM
  if (tid == 0) {
    // 2D tensor: dim0 = L elements (2B each), dim1 = B rows, row stride DM*L elems.
    const unsigned long long ga =
        (unsigned long long)(uintptr_t)(Xbf + (size_t)d * LL);
    const unsigned int lds_off = (unsigned int)(uintptr_t)&ldsx[0];
    v4u g0 = { 0x1u,                                   // count=1, no gather
               lds_off,                                 // lds_addr (bytes)
               (unsigned int)ga,                        // global_addr[31:0]
               (unsigned int)(ga >> 32) | 0x80000000u };// addr[56:32] | type=2
    v8i g1 = { (int)(1u << 16),        // data_size=1 (2 bytes), no multicast
               (int)(2048u << 16),     // tensor_dim0[15:0]=2048 in bits 63:48
               (int)(4u << 16),        // tensor_dim0 hi=0, tensor_dim1=4
               (int)(2048u << 16),     // tensor_dim1 hi=0, tile_dim0=2048
               4,                      // tile_dim1=4, tile_dim2=0
               0x200000,               // tensor_dim0_stride = DM*LL = 2097152
               0, 0 };                 // stride hi / tensor_dim1_stride = 0
    v4i g2 = { 0, 0, 0, 0 };
    v4i g3 = { 0, 0, 0, 0 };
    v8i g4 = { 0, 0, 0, 0, 0, 0, 0, 0 };
    __builtin_amdgcn_tensor_load_to_lds(g0, g1, g2, g3, g4, 0);
    __builtin_amdgcn_s_wait_tensorcnt((short)0);
  }
#else
  for (int i = tid; i < 4 * LL; i += 256) {
    const int b = i >> 11, l = i & (LL - 1);
    ldsx[i] = Xbf[(((size_t)(b << 10) + d) << 11) + l];
  }
#endif
  for (int i = tid; i < LL; i += 256) ldsx[4 * LL + i] = 0;  // zero row pads N to 16
  __syncthreads();

  const int wave = tid >> 5;
  const int lane = tid & 31;
  const int n = lane & 15;                 // A row index / B col (batch) index
  const int h = lane >> 4;
  const int xrow = (n < 4) ? n : 4;        // batches >= 4 read the zero row
  const unsigned short* xb = &ldsx[xrow * LL];
  const float sk = skip[d];
  const v8f vzero = {0.f,0.f,0.f,0.f,0.f,0.f,0.f,0.f};

  for (int lt = wave; lt < (LL / 16); lt += 8) {
    const int l0 = lt << 4;
    v8f acc = vzero;
    const int nsteps = ((l0 + 15) >> 5) + 1;   // causal: m tiles covering [0, l0+15]

    for (int s = 0; s < nsteps; ++s) {
      const int m0 = s << 5;
      // B fragment: X[m0+k][batch], 16 contiguous bf16 per lane (aligned b128 ds loads)
      Frag fb;
      const unsigned short* pb = xb + m0 + (h << 4);
      fb.q[0] = *(const u32x4*)pb;
      fb.q[1] = *(const u32x4*)(pb + 8);
      // A fragment: A[i][j] = ker[l0-m0 + i - j] = rker[(L-1)-(l0-m0)-i + j]
      // pre-paired dwords -> one ds_load_b32 per fragment VGPR, any alignment.
      const int base = (LL - 1) - (l0 - m0) - n + (h << 4);
      Frag fa;
      #pragma unroll
      for (int v = 0; v < 4; ++v) {
        fa.u[v]     = rker32[base + 2 * v];
        fa.u[4 + v] = rker32[base + 8 + 2 * v];
      }
      acc = wmma_bf16(fa.bf, fb.bf, acc);
    }

    // Fused skip + gate epilogue (only batch lanes 0..3 are real).
    if (n < 4) {
      const int lbase = l0 + (h << 3);
      const size_t rowoff = (((size_t)(n << 10) + d) << 11) + lbase; // (b*DM+d)*L + l
      if (mode == 0) {
        #pragma unroll
        for (int v = 0; v < 8; ++v) {
          const float kv = (acc[v] + sk * X[rowoff + v]) * gate[rowoff + v];
          outA[rowoff + v]   = kv;
          outAbf[rowoff + v] = f2bf(kv);
        }
      } else {
        const size_t colbase = ((size_t)n << 11) + lbase;            // b*L + l
        #pragma unroll
        for (int v = 0; v < 8; ++v) {
          const float y = (acc[v] + sk * X[rowoff + v]) * gate[rowoff + v];
          outB[((colbase + v) << 10) + d] = f2bf(y);                 // (bl, h) bf16
        }
      }
    }
  }
}

// ---------------- host-side orchestration ----------------
extern "C" void kernel_launch(void* const* d_in, const int* in_sizes, int n_in,
                              void* d_out, int out_size, void* d_ws, size_t ws_size,
                              hipStream_t stream) {
  (void)in_sizes; (void)n_in; (void)out_size; (void)ws_size;

  const float* u   = (const float*)d_in[0];
  const float* Wq  = (const float*)d_in[1];
  const float* bq  = (const float*)d_in[2];
  const float* Wk  = (const float*)d_in[3];
  const float* bk  = (const float*)d_in[4];
  const float* Wv  = (const float*)d_in[5];
  const float* bv  = (const float*)d_in[6];
  const float* kk  = (const float*)d_in[7];   // k_kernel (DM, L)
  const float* kD  = (const float*)d_in[8];   // k_D (DM)
  const float* ssm = (const float*)d_in[9];   // ssm_kernel (DM, L)
  const float* Dsk = (const float*)d_in[10];  // D (DM)
  const float* Wo  = (const float*)d_in[11];
  const float* bo  = (const float*)d_in[12];
  float* outp = (float*)d_out;

  // workspace layout (bytes)
  char* ws = (char*)d_ws;
  const size_t NBDL = (size_t)BB * DM * LL;                 // 8388608 elements
  float*          qbuf = (float*)(ws + 0);                  // 33554432 B (b,d,l) fp32
  float*          kbuf = (float*)(ws + 33554432);           // 33554432 B k, then kv
  float*          vbuf = (float*)(ws + 67108864);           // 33554432 B
  unsigned short* ubf  = (unsigned short*)(ws + 100663296); // 16777216 B u bf16 (bl,h)
  unsigned short* ybf  = (unsigned short*)(ws + 117440512); // 16777216 B y bf16 (bl,h)
  unsigned short* wqb  = (unsigned short*)(ws + 134217728); // 2 MB each
  unsigned short* wkb  = (unsigned short*)(ws + 136314880);
  unsigned short* wvb  = (unsigned short*)(ws + 138412032);
  unsigned short* wob  = (unsigned short*)(ws + 140509184);
  unsigned short* kbf  = (unsigned short*)(ws + 142606336); // 16777216 B k/kv bf16 (b,d,l)

  // 0) one-time bf16 conversions (weights stay L2-resident afterwards)
  cvt_bf16<<<(unsigned)((NBDL + 255) / 256), 256, 0, stream>>>(u,  ubf, (int)NBDL);
  cvt_bf16<<<(DM * DM + 255) / 256, 256, 0, stream>>>(Wq, wqb, DM * DM);
  cvt_bf16<<<(DM * DM + 255) / 256, 256, 0, stream>>>(Wk, wkb, DM * DM);
  cvt_bf16<<<(DM * DM + 255) / 256, 256, 0, stream>>>(Wv, wvb, DM * DM);
  cvt_bf16<<<(DM * DM + 255) / 256, 256, 0, stream>>>(Wo, wob, DM * DM);

  // 1) QKV projections: (8192 x 1024) x (1024 x 1024)^T, stored transposed (b,d,l)
  //    k also emits a bf16 shadow for the TDM-staged conv input.
  dim3 gg((BB * LL) / 32, DM / 32);
  gemm_bf16<<<gg, 32, 0, stream>>>(ubf, wqb, bq, qbuf, nullptr, 0);
  gemm_bf16<<<gg, 32, 0, stream>>>(ubf, wkb, bk, kbuf, kbf,     0);
  gemm_bf16<<<gg, 32, 0, stream>>>(ubf, wvb, bv, vbuf, nullptr, 0);

  // 2) shift conv on k, + k_D skip, * v -> kv (fp32+bf16, in place over k buffers)
  conv_gate<<<DM, 256, 0, stream>>>(kk, kbuf, kbf, kD, vbuf, kbuf, kbf, nullptr, 0);

  // 3) ssm conv on kv, + D skip, * q -> y bf16, transposed to (bl, h)
  conv_gate<<<DM, 256, 0, stream>>>(ssm, kbuf, kbf, Dsk, qbuf, nullptr, nullptr, ybf, 1);

  // 4) output projection: out = y @ Wo^T + bo, row-major (b, l, o)
  gemm_bf16<<<gg, 32, 0, stream>>>(ybf, wob, bo, outp, nullptr, 1);
}
